// NumberSequenceEncoder_60026462929333
// MI455X (gfx1250) — matrix operand order repaired and checked by reading, not compiled, not run on hardware
//
#include <hip/hip_runtime.h>

typedef __attribute__((ext_vector_type(16))) __bf16 v16bf;
typedef __attribute__((ext_vector_type(8)))  float  v8f;

#define EMB      512
#define GATES    2048      // 4*EMB
#define SEQ      64
#define BATCH    2048
#define ROW_PAD  520       // LDS row stride in ushorts (512 + 8 pad, keeps 16B alignment)
#define NKT      16        // K tiles per GEMM (512 / 32)

union Frag {
    v16bf v;
    uint4 q[2];
};

static __device__ __forceinline__ unsigned short f32_to_bf16(float f) {
    unsigned u = __builtin_bit_cast(unsigned, f);
    u += 0x7FFFu + ((u >> 16) & 1u);   // round-to-nearest-even
    return (unsigned short)(u >> 16);
}

static __device__ __forceinline__ float sigmoidf_fast(float x) {
    return 1.0f / (1.0f + __expf(-x));          // exp(-x)->inf gives 0: safe
}
static __device__ __forceinline__ float tanhf_fast(float x) {
    return 2.0f / (1.0f + __expf(-2.0f * x)) - 1.0f;  // overflow-safe
}

// ---------------------------------------------------------------------------
// Pre-swizzle fp32 weights [2048][512] row-major into bf16 WMMA-B fragment
// order: tile (nt,kt) of 32x16 (KxN) is 512 contiguous bf16; lane l's 16
// elements are contiguous at lane*16.  lane = (k%32)/16*16 + n%16,
// element = k%16.
// ---------------------------------------------------------------------------
__global__ void swizzle_weights(const float* __restrict__ w,
                                unsigned short* __restrict__ out) {
    int idx = blockIdx.x * blockDim.x + threadIdx.x;
    if (idx >= GATES * EMB) return;
    int n  = idx >> 9;          // row of W (gate output index)
    int k  = idx & 511;         // col of W (input feature)
    int nt = n >> 4, ni = n & 15;
    int kt = k >> 5, kk = k & 31;
    int half = kk >> 4, rem = kk & 15;
    int lane = half * 16 + ni;
    int o = ((nt * NKT + kt) << 9) + lane * 16 + rem;
    out[o] = f32_to_bf16(w[idx]);
}

__global__ void combine_bias(const float* __restrict__ b_ih,
                             const float* __restrict__ b_hh,
                             float* __restrict__ bias) {
    int i = blockIdx.x * blockDim.x + threadIdx.x;
    if (i < GATES) bias[i] = b_ih[i] + b_hh[i];
}

// ---------------------------------------------------------------------------
// Persistent fused LSTM kernel.  Each block owns 16 batch rows for the whole
// sequence; 8 waves x 4 e-tiles each cover the 512 hidden dims.  Per step:
//   gather x_t -> LDS (bf16), gates = x@Wih^T + h@Whh^T + b via bf16 WMMA,
//   elementwise LSTM cell in fp32 registers, h -> LDS bf16 for next step.
// No inter-block communication needed.
// ---------------------------------------------------------------------------
__global__ __launch_bounds__(256)
void lstm_persist(const int* __restrict__ x,
                  const float* __restrict__ emb,
                  const unsigned short* __restrict__ wih,   // swizzled bf16
                  const unsigned short* __restrict__ whh,   // swizzled bf16
                  const float* __restrict__ bias,
                  float* __restrict__ out) {
    __shared__ unsigned short lx[16 * ROW_PAD];   // x_t tile, bf16
    __shared__ unsigned short lh[16 * ROW_PAD];   // h tile, bf16

    const int tid  = threadIdx.x;
    const int wave = tid >> 5;
    const int lane = tid & 31;
    const int half = lane >> 4;
    const int ni   = lane & 15;
    const int bb   = blockIdx.x * 16;   // batch row base

    // zero initial hidden state
    for (int i = tid; i < 16 * ROW_PAD; i += 256) lh[i] = 0;

    float c_reg[4][8];
    float h_reg[4][8];
#pragma unroll
    for (int j = 0; j < 4; ++j)
#pragma unroll
        for (int r = 0; r < 8; ++r) { c_reg[j][r] = 0.0f; h_reg[j][r] = 0.0f; }

    // preload per-lane bias values: acc init (same value for all 8 rows)
    float bias_r[4][4];
#pragma unroll
    for (int j = 0; j < 4; ++j) {
        int et = wave * 4 + j;
#pragma unroll
        for (int q = 0; q < 4; ++q)
            bias_r[j][q] = bias[q * EMB + et * 16 + ni];
    }

    // gather mapping: 16 threads per batch row, 32 floats each
    const int gm = tid >> 4;
    const int gp = tid & 15;

    for (int t = 0; t < SEQ; ++t) {
        __syncthreads();   // prev-step LDS reads & h writes all complete

        // ---- gather x_t row -> LDS bf16 ----
        {
            int idx = x[(bb + gm) * SEQ + t];
            const float4* src = (const float4*)(emb + (long)idx * EMB) + gp * 8;
#pragma unroll
            for (int i = 0; i < 8; ++i) {
                float4 v = src[i];
                ushort4 u;
                u.x = f32_to_bf16(v.x);
                u.y = f32_to_bf16(v.y);
                u.z = f32_to_bf16(v.z);
                u.w = f32_to_bf16(v.w);
                *(ushort4*)&lx[gm * ROW_PAD + gp * 32 + i * 4] = u;
            }
        }
        __syncthreads();   // x_t visible to all waves

        // ---- gates + cell update, 4 e-tiles per wave ----
#pragma unroll
        for (int j = 0; j < 4; ++j) {
            const int et = wave * 4 + j;
            v8f acc[4];
#pragma unroll
            for (int q = 0; q < 4; ++q)
#pragma unroll
                for (int r = 0; r < 8; ++r) acc[q][r] = bias_r[j][q];

#pragma unroll
            for (int s = 0; s < 2; ++s) {
                const unsigned short* A = s ? lh : lx;
                const uint4* W = (const uint4*)(s ? whh : wih);
#pragma unroll 4
                for (int kt = 0; kt < NKT; ++kt) {
                    Frag fa;
                    const int abase = ni * ROW_PAD + kt * 32 + half * 16;
                    fa.q[0] = *(const uint4*)&A[abase];
                    fa.q[1] = *(const uint4*)&A[abase + 8];
#pragma unroll
                    for (int q = 0; q < 4; ++q) {
                        const uint4* wp =
                            W + ((((q * 32 + et) * NKT + kt) << 6) + lane * 2);
                        Frag fb;
                        fb.q[0] = wp[0];
                        fb.q[1] = wp[1];
                        acc[q] = __builtin_amdgcn_wmma_f32_16x16x32_bf16(
                            false, fa.v, false, fb.v, (short)0, acc[q],
                            false, false);
                    }
                }
            }

            // elementwise LSTM cell (gate order: i, f, g, o)
#pragma unroll
            for (int r = 0; r < 8; ++r) {
                float iv = sigmoidf_fast(acc[0][r]);
                float fv = sigmoidf_fast(acc[1][r]);
                float gv = tanhf_fast(acc[2][r]);
                float ov = sigmoidf_fast(acc[3][r]);
                float cn = fv * c_reg[j][r] + iv * gv;
                c_reg[j][r] = cn;
                h_reg[j][r] = ov * tanhf_fast(cn);
            }
        }

        __syncthreads();   // all waves finished reading lh

        // ---- write new h (bf16) for next step's recurrent GEMM ----
#pragma unroll
        for (int j = 0; j < 4; ++j) {
            const int et = wave * 4 + j;
#pragma unroll
            for (int r = 0; r < 8; ++r) {
                int row = r + half * 8;
                lh[row * ROW_PAD + et * 16 + ni] = f32_to_bf16(h_reg[j][r]);
            }
        }
    }

    // ---- final h -> global fp32 ----
#pragma unroll
    for (int j = 0; j < 4; ++j) {
        const int et = wave * 4 + j;
#pragma unroll
        for (int r = 0; r < 8; ++r) {
            int row = r + half * 8;
            out[(long)(bb + row) * EMB + et * 16 + ni] = h_reg[j][r];
        }
    }
}

extern "C" void kernel_launch(void* const* d_in, const int* in_sizes, int n_in,
                              void* d_out, int out_size, void* d_ws, size_t ws_size,
                              hipStream_t stream) {
    (void)in_sizes; (void)n_in; (void)out_size; (void)ws_size;
    const int*   x    = (const int*)d_in[0];
    const float* emb  = (const float*)d_in[1];
    const float* W_ih = (const float*)d_in[2];
    const float* W_hh = (const float*)d_in[3];
    const float* b_ih = (const float*)d_in[4];
    const float* b_hh = (const float*)d_in[5];
    float* out = (float*)d_out;

    unsigned short* wih_s = (unsigned short*)d_ws;                 // 2 MB
    unsigned short* whh_s = wih_s + (size_t)GATES * EMB;           // 2 MB
    float* bias = (float*)(whh_s + (size_t)GATES * EMB);           // 8 KB

    const int nW = GATES * EMB;
    swizzle_weights<<<(nW + 255) / 256, 256, 0, stream>>>(W_ih, wih_s);
    swizzle_weights<<<(nW + 255) / 256, 256, 0, stream>>>(W_hh, whh_s);
    combine_bias<<<(GATES + 255) / 256, 256, 0, stream>>>(b_ih, b_hh, bias);

    lstm_persist<<<BATCH / 16, 256, 0, stream>>>(x, emb, wih_s, whh_s, bias, out);
}
